// SchNet_37220186587161
// MI455X (gfx1250) — matrix operand order repaired and checked by reading, not compiled, not run on hardware
//
#include <hip/hip_runtime.h>
#include <hip/hip_bf16.h>
#include <math.h>

typedef __attribute__((ext_vector_type(16))) _Float16 v16h;
typedef __attribute__((ext_vector_type(8)))  _Float16 v8h;
typedef __attribute__((ext_vector_type(8)))  float    v8f;

#define NB   2048
#define AT   64
#define NTOT (NB*AT)
#define DD   128
#define FF   128
#define LL   3
#define WS   136   // padded LDS row stride in halves: 272B = 4-bank step -> conflict-free frags

// ---------------- distance + cutoff kernel ----------------
__global__ void dist_kernel(const float* __restrict__ pos,
                            float* __restrict__ dmean,
                            float* __restrict__ cutv) {
    __shared__ float px[AT], py[AT], pz[AT];
    const int b = blockIdx.x, i = threadIdx.x;
    const int base = b * AT;
    px[i] = pos[(base + i) * 3 + 0];
    py[i] = pos[(base + i) * 3 + 1];
    pz[i] = pos[(base + i) * 3 + 2];
    __syncthreads();
    const float xi = px[i], yi = py[i], zi = pz[i];
    float s = 0.0f;
#pragma unroll 8
    for (int j = 0; j < AT; ++j) {
        float dx = xi - px[j], dy = yi - py[j], dz = zi - pz[j];
        s += sqrtf(dx * dx + dy * dy + dz * dz + 1e-12f);
    }
    const float dm = s * (1.0f / AT);
    dmean[base + i] = dm;
    cutv[base + i] = (dm <= 5.0f) ? cosf(dm * 0.3141592653589793f) : 0.0f;
}

// ---------------- fully fused SchNet kernel ----------------
__launch_bounds__(128)
__global__ void schnet_kernel(const int*   __restrict__ Z,
                              const float* __restrict__ emb,
                              const float* __restrict__ fw1,
                              const float* __restrict__ fb1,
                              const float* __restrict__ fw2,
                              const float* __restrict__ fb2,
                              const float* __restrict__ cw,
                              const float* __restrict__ cb,
                              const float* __restrict__ dw1,
                              const float* __restrict__ db1,
                              const float* __restrict__ dw2,
                              const float* __restrict__ db2,
                              const float* __restrict__ ow1,
                              const float* __restrict__ ob1,
                              const float* __restrict__ ow2,
                              const float* __restrict__ ob2,
                              const float* __restrict__ dmean,
                              const float* __restrict__ cutv,
                              float* __restrict__ out) {
    __shared__ _Float16 wbuf[128 * WS];        // transposed weights WT[n][k], f16
    __shared__ _Float16 abuf[4][16 * WS];      // per-wave activation tile [m][k], f16
    __shared__ float    s_fw1[FF], s_fb1[FF];

    const int tid  = threadIdx.x;
    const int wave = tid >> 5;
    const int lane = tid & 31;
    const int half = lane >> 4;   // lane-half (K/M split per ISA layouts)
    const int ln   = lane & 15;
    const int tile_base = blockIdx.x * 64 + wave * 16;

    _Float16* aw = abuf[wave];

    // per-lane row scalars
    const float dA = dmean[tile_base + ln];    // A-frag row (M = ln)
    float cutr[8];
    int   rowc[8];
#pragma unroll
    for (int r = 0; r < 8; ++r) {
        rowc[r] = tile_base + r + half * 8;    // C-layout rows
        cutr[r] = cutv[rowc[r]];
    }

    // x accumulator in WMMA C layout: xacc[t][r] = x[row r+half*8][t*16+ln]
    v8f xacc[8];
#pragma unroll
    for (int r = 0; r < 8; ++r) {
        const float* er = emb + Z[rowc[r]] * DD;
#pragma unroll
        for (int t = 0; t < 8; ++t) xacc[t][r] = er[t * 16 + ln];
    }

    v8f czero;
#pragma unroll
    for (int r = 0; r < 8; ++r) czero[r] = 0.0f;

    // stage a row-major [K][2^ncshift] f32 weight into wbuf transposed (f16)
    auto stageW = [&](const float* W, int ncshift, int total) {
        __syncthreads();
        for (int idx = tid; idx < total; idx += 128) {
            int k = idx >> ncshift, n = idx & ((1 << ncshift) - 1);
            wbuf[n * WS + k] = (_Float16)W[idx];
        }
        __syncthreads();
    };

    // A fragments (16x32 f16 per kb) from this wave's activation tile
    auto loadA = [&](v16h* afr) {
        const _Float16* row = aw + ln * WS;
#pragma unroll
        for (int kb = 0; kb < 4; ++kb) {
            v8h lo = *(const v8h*)(row + kb * 32 + half * 8);
            v8h hi = *(const v8h*)(row + kb * 32 + 16 + half * 8);
            v16h a;
#pragma unroll
            for (int j = 0; j < 8; ++j) { a[j] = lo[j]; a[j + 8] = hi[j]; }
            afr[kb] = a;
        }
    };

    // B fragment (32x16) for output-col tile n0, k-block kb
    auto loadB = [&](int n0, int kb) -> v16h {
        const _Float16* p = wbuf + (n0 + ln) * WS + kb * 32 + half * 8;
        v8h lo = *(const v8h*)p;
        v8h hi = *(const v8h*)(p + 16);
        v16h b;
#pragma unroll
        for (int j = 0; j < 8; ++j) { b[j] = lo[j]; b[j + 8] = hi[j]; }
        return b;
    };

    // Full stage GEMM: C[t] += A(16x128) * W(128x16*ntiles), tile-interleaved so
    // consecutive WMMAs hit independent accumulators; B frags batched 4 tiles at
    // a time so one ds-wait covers a group and loads overlap WMMA execution.
    auto gemmAll = [&](const v16h* afr, v8f* c, int ntiles) {
#pragma unroll
        for (int kb = 0; kb < 4; ++kb) {
#pragma unroll
            for (int tg = 0; tg < ntiles; tg += 4) {
                v16h b0 = loadB((tg + 0) * 16, kb);
                v16h b1 = loadB((tg + 1) * 16, kb);
                v16h b2 = loadB((tg + 2) * 16, kb);
                v16h b3 = loadB((tg + 3) * 16, kb);
                c[tg + 0] = __builtin_amdgcn_wmma_f32_16x16x32_f16(false, afr[kb], false, b0,
                                                                   (short)0, c[tg + 0], false, false);
                c[tg + 1] = __builtin_amdgcn_wmma_f32_16x16x32_f16(false, afr[kb], false, b1,
                                                                   (short)0, c[tg + 1], false, false);
                c[tg + 2] = __builtin_amdgcn_wmma_f32_16x16x32_f16(false, afr[kb], false, b2,
                                                                   (short)0, c[tg + 2], false, false);
                c[tg + 3] = __builtin_amdgcn_wmma_f32_16x16x32_f16(false, afr[kb], false, b3,
                                                                   (short)0, c[tg + 3], false, false);
            }
        }
    };

    v16h afr[4];
    v8f  c[8];
    float bb[8];

    for (int l = 0; l < LL; ++l) {
        // ---- stage fw2 (and this layer's filter params) ----
        __syncthreads();
        s_fw1[tid] = fw1[l * FF + tid];
        s_fb1[tid] = fb1[l * FF + tid];
        {
            const float* W = fw2 + l * FF * FF;
            for (int idx = tid; idx < FF * FF; idx += 128) {
                int k = idx >> 7, n = idx & 127;
                wbuf[n * WS + k] = (_Float16)W[idx];
            }
        }
        __syncthreads();

        // f = relu(d*fw1+fb1), built directly as A fragments (rank-1 input)
#pragma unroll
        for (int kb = 0; kb < 4; ++kb) {
            v16h a;
#pragma unroll
            for (int j = 0; j < 16; ++j) {
                int k = kb * 32 + half * 8 + ((j < 8) ? j : (8 + j)); // j>=8 -> 16+(j-8)
                float fv = dA * s_fw1[k] + s_fb1[k];
                a[j] = (_Float16)(fv > 0.0f ? fv : 0.0f);
            }
            afr[kb] = a;
        }
        // ---- S1: h = (f @ fw2 + fb2) * cut[row] ----
        {
            const float* bias = fb2 + l * FF;
#pragma unroll
            for (int t = 0; t < 8; ++t) { bb[t] = bias[t * 16 + ln]; c[t] = czero; }
            gemmAll(afr, c, 8);
#pragma unroll
            for (int t = 0; t < 8; ++t)
#pragma unroll
                for (int r = 0; r < 8; ++r)
                    aw[(r + half * 8) * WS + t * 16 + ln] = (_Float16)((c[t][r] + bb[t]) * cutr[r]);
        }

        // ---- S2: v = h @ cw + cb ----
        stageW(cw + l * FF * DD, 7, FF * DD);
        loadA(afr);
        {
            const float* bias = cb + l * DD;
#pragma unroll
            for (int t = 0; t < 8; ++t) { bb[t] = bias[t * 16 + ln]; c[t] = czero; }
            gemmAll(afr, c, 8);
#pragma unroll
            for (int t = 0; t < 8; ++t)
#pragma unroll
                for (int r = 0; r < 8; ++r)
                    aw[(r + half * 8) * WS + t * 16 + ln] = (_Float16)(c[t][r] + bb[t]);
        }

        // ---- S3: u = relu(v @ dw1 + db1) ----
        stageW(dw1 + l * DD * DD, 7, DD * DD);
        loadA(afr);
        {
            const float* bias = db1 + l * DD;
#pragma unroll
            for (int t = 0; t < 8; ++t) { bb[t] = bias[t * 16 + ln]; c[t] = czero; }
            gemmAll(afr, c, 8);
#pragma unroll
            for (int t = 0; t < 8; ++t)
#pragma unroll
                for (int r = 0; r < 8; ++r) {
                    float v = c[t][r] + bb[t];
                    aw[(r + half * 8) * WS + t * 16 + ln] = (_Float16)(v > 0.0f ? v : 0.0f);
                }
        }

        // ---- S4: x += u @ dw2 + db2  (residual via WMMA C operand) ----
        stageW(dw2 + l * DD * DD, 7, DD * DD);
        loadA(afr);
        {
            const float* bias = db2 + l * DD;
#pragma unroll
            for (int t = 0; t < 8; ++t) bb[t] = bias[t * 16 + ln];
            gemmAll(afr, xacc, 8);
#pragma unroll
            for (int t = 0; t < 8; ++t)
#pragma unroll
                for (int r = 0; r < 8; ++r) xacc[t][r] += bb[t];
        }
    }

    // ---- output net: relu(x @ ow1 + ob1) @ ow2 + ob2 ----
#pragma unroll
    for (int t = 0; t < 8; ++t)
#pragma unroll
        for (int r = 0; r < 8; ++r)
            aw[(r + half * 8) * WS + t * 16 + ln] = (_Float16)xacc[t][r];

    stageW(ow1, 6, DD * 64);   // 128 x 64
    loadA(afr);

    float p[8];
#pragma unroll
    for (int r = 0; r < 8; ++r) p[r] = 0.0f;
    const float ob2v = ob2[0];
    float b1v[4], w2v[4];
#pragma unroll
    for (int t = 0; t < 4; ++t) {
        b1v[t] = ob1[t * 16 + ln];
        w2v[t] = ow2[t * 16 + ln];
        c[t] = czero;
    }
    gemmAll(afr, c, 4);
#pragma unroll
    for (int t = 0; t < 4; ++t)
#pragma unroll
        for (int r = 0; r < 8; ++r) {
            float u = c[t][r] + b1v[t];
            u = u > 0.0f ? u : 0.0f;
            p[r] += u * w2v[t];
        }
    // reduce the 16 N-lanes within each lane-half (masks < 16 stay in-half)
#pragma unroll
    for (int r = 0; r < 8; ++r) {
        float v = p[r];
        v += __shfl_xor(v, 1);
        v += __shfl_xor(v, 2);
        v += __shfl_xor(v, 4);
        v += __shfl_xor(v, 8);
        if (ln == 0) out[rowc[r]] = v + ob2v;
    }
}

// ---------------- launch ----------------
extern "C" void kernel_launch(void* const* d_in, const int* in_sizes, int n_in,
                              void* d_out, int out_size, void* d_ws, size_t ws_size,
                              hipStream_t stream) {
    (void)in_sizes; (void)n_in; (void)out_size; (void)ws_size;
    const int*   Z    = (const int*)  d_in[0];
    const float* pos  = (const float*)d_in[1];
    // d_in[2] batch_idx: implied by contiguous equal-size blocks (as in reference)
    const float* emb  = (const float*)d_in[3];
    const float* fw1  = (const float*)d_in[4];
    const float* fb1  = (const float*)d_in[5];
    const float* fw2  = (const float*)d_in[6];
    const float* fb2  = (const float*)d_in[7];
    const float* cw   = (const float*)d_in[8];
    const float* cb   = (const float*)d_in[9];
    const float* dw1  = (const float*)d_in[10];
    const float* db1  = (const float*)d_in[11];
    const float* dw2  = (const float*)d_in[12];
    const float* db2  = (const float*)d_in[13];
    const float* ow1  = (const float*)d_in[14];
    const float* ob1  = (const float*)d_in[15];
    const float* ow2  = (const float*)d_in[16];
    const float* ob2  = (const float*)d_in[17];
    float* out   = (float*)d_out;
    float* dmean = (float*)d_ws;
    float* cutv  = dmean + NTOT;

    dist_kernel<<<NB, AT, 0, stream>>>(pos, dmean, cutv);
    schnet_kernel<<<NTOT / 64, 128, 0, stream>>>(Z, emb, fw1, fb1, fw2, fb2, cw, cb,
                                                 dw1, db1, dw2, db2, ow1, ob1, ow2, ob2,
                                                 dmean, cutv, out);
}
